// ImageGraphClip_44933947851323
// MI455X (gfx1250) — compile-verified, matrix operand order checked by reading, not run on hardware
//
// CLIP-style ResNet+GraphTransformer forward for MI455X (gfx1250), wave32 + WMMA bf16.
// All matmul-shaped work (convs via implicit im2col, 1x1 convs, all transformer
// projections, attnpool projections) runs on v_wmma_f32_16x16x32_bf16 with f32
// accumulate; BN/bias/residual/ReLU/quickGELU fused into GEMM epilogues.
// Activations kept NHWC so 1x1 convs are pure GEMMs and channel epilogues are per-N.
// This revision: double-buffered LDS pipeline (global loads overlap WMMA) and
// branch-free clamped global loads (no saveexec/waitcnt serialization in loaders).
#include <hip/hip_runtime.h>
#include <cmath>

typedef __attribute__((ext_vector_type(16))) __bf16 bf16x16;
typedef __attribute__((ext_vector_type(8)))  __bf16 bf16x8;
typedef __attribute__((ext_vector_type(4)))  __bf16 bf16x4;
typedef __attribute__((ext_vector_type(8)))  float  f32x8;

#define BMT 128
#define BNT 128
#define KTT 32
#define LDT 40   // padded LDS row (bf16 elems) to stagger banks

// ---- fragment load from LDS, per ISA 7.12.2 (16-bit A 16x32 layout) ----
__device__ __forceinline__ bf16x16 frag_ld(const __bf16* tile, int r0, int lane) {
  int row = r0 + (lane & 15);
  int grp = (lane >> 4) << 3;                   // lanes 0-15: K 0..7/16..23 ; 16-31: 8..15/24..31
  const __bf16* p = tile + row * LDT + grp;
  union { bf16x16 v; bf16x8 h[2]; } u;
  u.h[0] = *(const bf16x8*)(p);
  u.h[1] = *(const bf16x8*)(p + 16);
  return u.v;
}

__device__ __forceinline__ void epilogue_store(
    f32x8 (&acc)[4][2], float* __restrict__ C, int M, int N,
    int m0, int n0, int wm, int wn, int lane,
    const float* scale, float smul, const float* bias,
    const float* add, int flags)
{
#pragma unroll
  for (int j = 0; j < 2; ++j) {
    int n = n0 + wn + j * 16 + (lane & 15);
    if (n >= N) continue;
    float sc = scale ? scale[n] * smul : 1.0f;
    float bs = bias ? bias[n] : 0.0f;
#pragma unroll
    for (int i = 0; i < 4; ++i) {
      int mb = m0 + wm + i * 16 + ((lane >> 4) << 3);
#pragma unroll
      for (int v = 0; v < 8; ++v) {
        int m = mb + v;
        if (m >= M) continue;
        float val = acc[i][j][v] * sc + bs;
        if (add) val += add[(size_t)m * N + n];
        if (flags & 1) val = fmaxf(val, 0.0f);                       // ReLU
        if (flags & 2) val = val / (1.0f + __expf(-1.702f * val));   // quickGELU
        C[(size_t)m * N + n] = val;
      }
    }
  }
}

// ---- dense GEMM: C[M,N] = A[M,K](lda) * B[N,K]^T, f32 in/out, bf16 WMMA core ----
// Requires K % 32 == 0 (true for every dense call site in this network).
__global__ __launch_bounds__(256) void gemm_bf16_wmma(
    const float* __restrict__ A, int lda, const float* __restrict__ B,
    float* __restrict__ C, int M, int N, int K,
    const float* __restrict__ scale, float smul,
    const float* __restrict__ bias, const float* __restrict__ add, int flags)
{
  __shared__ __bf16 lsA[2][BMT * LDT];
  __shared__ __bf16 lsB[2][BNT * LDT];
  int m0 = blockIdx.x * BMT, n0 = blockIdx.y * BNT;
  int tid = threadIdx.x, lane = tid & 31, wave = tid >> 5;
  int wm = (wave & 1) * 64, wn = (wave >> 1) * 32;
  f32x8 acc[4][2] = {};
  int r = tid >> 1;                    // 2 threads per tile row
  int kc0 = (tid & 1) * 16;
  int gm = m0 + r; bool rok = gm < M;
  int gn = n0 + r; bool nok = gn < N;
  // Clamped base pointers: loads are always in-bounds; OOB rows zeroed by select.
  const float* Abase = A + (size_t)(rok ? gm : (M - 1)) * lda + kc0;
  const float* Bbase = B + (size_t)(nok ? gn : (N - 1)) * K + kc0;

  float4 ra[4], rb[4];
  auto gload = [&](int kt) {
#pragma unroll
    for (int q = 0; q < 4; ++q) {
      ra[q] = *(const float4*)(Abase + kt + q * 4);
      rb[q] = *(const float4*)(Bbase + kt + q * 4);
    }
    if (!rok) {
#pragma unroll
      for (int q = 0; q < 4; ++q) ra[q] = make_float4(0.f, 0.f, 0.f, 0.f);
    }
    if (!nok) {
#pragma unroll
      for (int q = 0; q < 4; ++q) rb[q] = make_float4(0.f, 0.f, 0.f, 0.f);
    }
  };
  auto lstore = [&](int p) {
#pragma unroll
    for (int q = 0; q < 4; ++q) {
      bf16x4 ba, bb;
      ba[0] = (__bf16)ra[q].x; ba[1] = (__bf16)ra[q].y;
      ba[2] = (__bf16)ra[q].z; ba[3] = (__bf16)ra[q].w;
      bb[0] = (__bf16)rb[q].x; bb[1] = (__bf16)rb[q].y;
      bb[2] = (__bf16)rb[q].z; bb[3] = (__bf16)rb[q].w;
      *(bf16x4*)(&lsA[p][r * LDT + kc0 + q * 4]) = ba;
      *(bf16x4*)(&lsB[p][r * LDT + kc0 + q * 4]) = bb;
    }
  };

  int nk = K / KTT;
  gload(0);
  int p = 0;
  for (int it = 0; it < nk; ++it) {
    lstore(p);
    __syncthreads();
    if (it + 1 < nk) gload((it + 1) * KTT);   // overlaps with WMMA below
    bf16x16 fa[4], fb[2];
#pragma unroll
    for (int i = 0; i < 4; ++i) fa[i] = frag_ld(lsA[p], wm + i * 16, lane);
#pragma unroll
    for (int j = 0; j < 2; ++j) fb[j] = frag_ld(lsB[p], wn + j * 16, lane);
#pragma unroll
    for (int i = 0; i < 4; ++i)
#pragma unroll
      for (int j = 0; j < 2; ++j)
        acc[i][j] = __builtin_amdgcn_wmma_f32_16x16x32_bf16(
            false, fa[i], false, fb[j], (short)0, acc[i][j], false, false);
    p ^= 1;
  }
  epilogue_store(acc, C, M, N, m0, n0, wm, wn, lane, scale, smul, bias, add, flags);
}

// ---- implicit-GEMM conv: Y[N*OH*OW, Oc] = im2col(X) * Wr[Oc, KH*KW*Ci]^T ----
__global__ __launch_bounds__(256) void conv_bf16_wmma(
    const float* __restrict__ X, int nchw, const float* __restrict__ Wr,
    float* __restrict__ Y, int Nn, int Hh, int Ww, int Ci, int Oc,
    int KH, int KW, int strd, int pad, int OH, int OW,
    const float* __restrict__ scale, float smul,
    const float* __restrict__ bias, const float* __restrict__ add, int flags)
{
  __shared__ __bf16 lsA[2][BMT * LDT];
  __shared__ __bf16 lsB[2][BNT * LDT];
  const int M = Nn * OH * OW;
  const int K = KH * KW * Ci;
  int m0 = blockIdx.x * BMT, n0 = blockIdx.y * BNT;
  int tid = threadIdx.x, lane = tid & 31, wave = tid >> 5;
  int wm = (wave & 1) * 64, wn = (wave >> 1) * 32;
  f32x8 acc[4][2] = {};
  int r = tid >> 1;
  int kc0 = (tid & 1) * 16;
  int gm = m0 + r; bool rok = gm < M;
  int ox = 0, oy = 0, nb = 0;
  if (rok) { ox = gm % OW; int t2 = gm / OW; oy = t2 % OH; nb = t2 / OH; }
  int gn = n0 + r; bool nok = gn < Oc;
  const float* Wbase = Wr + (size_t)(nok ? gn : (Oc - 1)) * K;

  float ra[16], rw[16];
  auto gload = [&](int kt) {
#pragma unroll 4
    for (int q = 0; q < 16; ++q) {
      int gk = kt + kc0 + q;
      int cgk = gk < K ? gk : K - 1;
      // weights: unconditional clamped load + select
      float wv = Wbase[cgk];
      rw[q] = (nok && gk < K) ? wv : 0.f;
      // im2col element (padding predicate is inherent)
      float va = 0.f;
      if (rok && gk < K) {
        int c = gk % Ci; int t3 = gk / Ci; int kx = t3 % KW; int ky = t3 / KW;
        int iy = oy * strd - pad + ky, ix = ox * strd - pad + kx;
        if (iy >= 0 && iy < Hh && ix >= 0 && ix < Ww)
          va = nchw ? X[(((size_t)nb * Ci + c) * Hh + iy) * Ww + ix]
                    : X[(((size_t)nb * Hh + iy) * Ww + ix) * Ci + c];
      }
      ra[q] = va;
    }
  };
  auto lstore = [&](int p) {
#pragma unroll
    for (int q = 0; q < 16; ++q) {
      lsA[p][r * LDT + kc0 + q] = (__bf16)ra[q];
      lsB[p][r * LDT + kc0 + q] = (__bf16)rw[q];
    }
  };

  int nk = (K + KTT - 1) / KTT;
  gload(0);
  int p = 0;
  for (int it = 0; it < nk; ++it) {
    lstore(p);
    __syncthreads();
    if (it + 1 < nk) gload((it + 1) * KTT);
    bf16x16 fa[4], fb[2];
#pragma unroll
    for (int i = 0; i < 4; ++i) fa[i] = frag_ld(lsA[p], wm + i * 16, lane);
#pragma unroll
    for (int j = 0; j < 2; ++j) fb[j] = frag_ld(lsB[p], wn + j * 16, lane);
#pragma unroll
    for (int i = 0; i < 4; ++i)
#pragma unroll
      for (int j = 0; j < 2; ++j)
        acc[i][j] = __builtin_amdgcn_wmma_f32_16x16x32_bf16(
            false, fa[i], false, fb[j], (short)0, acc[i][j], false, false);
    p ^= 1;
  }
  epilogue_store(acc, Y, M, Oc, m0, n0, wm, wn, lane, scale, smul, bias, add, flags);
}

// ---- OIHW -> [O, (ky,kx,c)] weight reorder ----
__global__ void k_wreorder(const float* __restrict__ w, float* __restrict__ out,
                           int O, int I, int KH, int KW) {
  int i = blockIdx.x * blockDim.x + threadIdx.x;
  int tot = O * I * KH * KW;
  if (i >= tot) return;
  int kw_ = i % KW; int t = i / KW; int kh_ = t % KH; t /= KH; int ic = t % I; int oc = t / I;
  out[((size_t)oc * KH * KW + kh_ * KW + kw_) * I + ic] = w[i];
}

// ---- 2x2 avgpool, NHWC ----
__global__ void k_avgpool2(const float* __restrict__ in, float* __restrict__ out,
                           int Nn, int H, int W, int C) {
  int OH = H >> 1, OW = W >> 1;
  size_t tot = (size_t)Nn * OH * OW * C;
  size_t i = (size_t)blockIdx.x * blockDim.x + threadIdx.x;
  if (i >= tot) return;
  int c = i % C; size_t t = i / C; int ox = t % OW; t /= OW; int oy = t % OH; int n = t / OH;
  const float* p = in + (((size_t)n * H + oy * 2) * W + ox * 2) * C + c;
  size_t rs = (size_t)W * C;
  out[i] = 0.25f * (p[0] + p[C] + p[rs] + p[rs + C]);
}

// ---- build attnpool tokens: tcat[b,0,:]=mean, tcat[b,1+j,:]=t[b,j,:], all +pos ----
__global__ void k_attn_tokens(const float* __restrict__ x, const float* __restrict__ pos,
                              float* __restrict__ tcat, int Bb, int C) {
  size_t tot = (size_t)Bb * 50 * C;
  size_t i = (size_t)blockIdx.x * blockDim.x + threadIdx.x;
  if (i >= tot) return;
  int c = i % C; size_t t = i / C; int j = t % 50; int b = t / 50;
  float v;
  if (j == 0) {
    float s = 0.f;
    for (int p2 = 0; p2 < 49; ++p2) s += x[((size_t)b * 49 + p2) * C + c];
    v = s * (1.0f / 49.0f);
  } else {
    v = x[((size_t)b * 49 + (j - 1)) * C + c];
  }
  tcat[i] = v + pos[(size_t)j * C + c];
}

// ---- attention pool core: 1 query x 50 keys, 8 heads, dh=256 ----
__global__ __launch_bounds__(256) void k_attnpool(const float* __restrict__ q,
                                                  const float* __restrict__ k,
                                                  const float* __restrict__ v,
                                                  float* __restrict__ out) {
  const int DH = 256, NP = 50, C = 2048, HE = 8;
  int b = blockIdx.x / HE, h = blockIdx.x % HE;
  __shared__ float sc[NP];
  int tid = threadIdx.x;
  const float* qp = q + (size_t)b * C + h * DH;
  if (tid < NP) {
    const float* kp = k + ((size_t)b * NP + tid) * C + h * DH;
    float s = 0.f;
    for (int d = 0; d < DH; ++d) s += qp[d] * kp[d];
    sc[tid] = s * 0.0625f;   // 1/sqrt(256)
  }
  __syncthreads();
  if (tid == 0) {
    float mx = -1e30f;
    for (int j = 0; j < NP; ++j) mx = fmaxf(mx, sc[j]);
    float sm = 0.f;
    for (int j = 0; j < NP; ++j) { float e = __expf(sc[j] - mx); sc[j] = e; sm += e; }
    float inv = 1.0f / sm;
    for (int j = 0; j < NP; ++j) sc[j] *= inv;
  }
  __syncthreads();
  if (tid < DH) {
    float o = 0.f;
    const float* vb = v + (size_t)b * NP * C + h * DH + tid;
    for (int j = 0; j < NP; ++j) o += sc[j] * vb[(size_t)j * C];
    out[(size_t)b * C + h * DH + tid] = o;
  }
}

// ---- masked graph attention: per (b,h,query-row), L=64 keys, dh=64 ----
__global__ __launch_bounds__(64) void k_graph_attn(const float* __restrict__ qkv,
                                                   const int* __restrict__ adj,
                                                   float* __restrict__ ao) {
  const int L = 64, E = 512, HE = 8, DH = 64, S3 = 1536;
  int blk = blockIdx.x;
  int qi = blk % L; int h = (blk / L) % HE; int b = blk / (L * HE);
  int tid = threadIdx.x;
  __shared__ float att[L];
  const float* qp = qkv + ((size_t)b * L + qi) * S3 + h * DH;
  const float* kp = qkv + ((size_t)b * L + tid) * S3 + 512 + h * DH;
  float s = 0.f;
  for (int d = 0; d < DH; ++d) s += qp[d] * kp[d];
  s *= 0.125f;                                   // 1/sqrt(64)
  if (adj[((size_t)b * L + qi) * L + tid] <= 0) s -= 1e9f;
  att[tid] = s;
  __syncthreads();
  float mx = -1e30f;
  for (int j = 0; j < L; ++j) mx = fmaxf(mx, att[j]);
  __syncthreads();
  float e = __expf(s - mx);
  att[tid] = e;
  __syncthreads();
  float sm = 0.f;
  for (int j = 0; j < L; ++j) sm += att[j];
  float o = 0.f;
  const float* vb = qkv + ((size_t)b * L) * S3 + 1024 + h * DH + tid;
  for (int j = 0; j < L; ++j) o += att[j] * vb[(size_t)j * S3];
  ao[((size_t)b * L + qi) * E + h * DH + tid] = o / sm;
}

// ---- LayerNorm over last dim ----
__global__ __launch_bounds__(256) void k_layernorm(const float* __restrict__ x,
                                                   float* __restrict__ y,
                                                   const float* __restrict__ g,
                                                   const float* __restrict__ b,
                                                   int dim) {
  int row = blockIdx.x;
  const float* xr = x + (size_t)row * dim;
  __shared__ float red[256];
  int tid = threadIdx.x;
  float s = 0.f;
  for (int i = tid; i < dim; i += 256) s += xr[i];
  red[tid] = s; __syncthreads();
  for (int o2 = 128; o2 > 0; o2 >>= 1) { if (tid < o2) red[tid] += red[tid + o2]; __syncthreads(); }
  float mean = red[0] / dim;
  __syncthreads();
  float vv = 0.f;
  for (int i = tid; i < dim; i += 256) { float d = xr[i] - mean; vv += d * d; }
  red[tid] = vv; __syncthreads();
  for (int o2 = 128; o2 > 0; o2 >>= 1) { if (tid < o2) red[tid] += red[tid + o2]; __syncthreads(); }
  float inv = rsqrtf(red[0] / dim + 1e-5f);
  for (int i = tid; i < dim; i += 256)
    y[(size_t)row * dim + i] = (xr[i] - mean) * inv * g[i] + b[i];
}

// ---- token embedding gather ----
__global__ void k_embed(const int* __restrict__ idx, const float* __restrict__ tok,
                        float* __restrict__ g, int rows, int E) {
  size_t i = (size_t)blockIdx.x * blockDim.x + threadIdx.x;
  if (i >= (size_t)rows * E) return;
  int e = i % E; int t = i / E;
  g[i] = tok[(size_t)idx[t] * E + e];
}

// ---- mean over L tokens ----
__global__ void k_mean_rows(const float* __restrict__ g, float* __restrict__ out,
                            int Bb, int L, int E) {
  int i = blockIdx.x * blockDim.x + threadIdx.x;
  if (i >= Bb * E) return;
  int e = i % E; int b = i / E;
  float s = 0.f;
  for (int l = 0; l < L; ++l) s += g[((size_t)b * L + l) * E + e];
  out[i] = s / (float)L;
}

// ---- per-row inverse L2 norm ----
__global__ __launch_bounds__(256) void k_row_inv_norm(const float* __restrict__ x,
                                                      float* __restrict__ inv, int dim) {
  int row = blockIdx.x;
  const float* xr = x + (size_t)row * dim;
  __shared__ float red[256];
  int tid = threadIdx.x;
  float s = 0.f;
  for (int i = tid; i < dim; i += 256) { float v = xr[i]; s += v * v; }
  red[tid] = s; __syncthreads();
  for (int o2 = 128; o2 > 0; o2 >>= 1) { if (tid < o2) red[tid] += red[tid + o2]; __syncthreads(); }
  if (tid == 0) inv[row] = 1.0f / fmaxf(sqrtf(red[0]), 1e-8f);
}

// ---- fused cosine similarity 32x32 output ----
__global__ __launch_bounds__(1024) void k_cosine(const float* __restrict__ a,
                                                 const float* __restrict__ ia,
                                                 const float* __restrict__ b,
                                                 const float* __restrict__ ib,
                                                 float* __restrict__ out) {
  int t = threadIdx.x;
  int i = t >> 5, j = t & 31;
  float s = 0.f;
  for (int e = 0; e < 512; ++e) s += a[(size_t)i * 512 + e] * b[(size_t)j * 512 + e];
  out[i * 32 + j] = s * ia[i] * ib[j];
}

extern "C" void kernel_launch(void* const* d_in, const int* in_sizes, int n_in,
                              void* d_out, int out_size, void* d_ws, size_t ws_size,
                              hipStream_t stream) {
  (void)in_sizes; (void)n_in; (void)out_size; (void)ws_size;
  auto F = [&](int i) { return (const float*)d_in[i]; };
  const float* image = F(0);
  const int* graphi = (const int*)d_in[1];
  const int* adj = (const int*)d_in[2];
  const float BN_SCALE = 1.0f / sqrtf(1.0f + 1e-5f);

  // Workspace bump allocator (assumes ws_size >= ~400 MB).
  float* W = (float*)d_ws;
  size_t off = 0;
  auto alloc = [&](size_t n) { float* p = W + off; off += n; return p; };
  float* ACT0 = alloc(25690112);   // full-res activations ping
  float* ACT1 = alloc(25690112);   // pong
  float* TMP0 = alloc(12845056);   // bottleneck intermediates
  float* TMP1 = alloc(12845056);
  float* WR   = alloc(2359296);    // reordered 3x3 weights
  float* TCAT = alloc(3276800);    // [32,50,2048]
  float* QP   = alloc(65536);
  float* KP   = alloc(3276800);
  float* VP   = alloc(3276800);
  float* ATTO = alloc(65536);
  float* IMGF = alloc(16384);
  float* Gb   = alloc(1048576);    // [2048,512] graph state
  float* Hb   = alloc(1048576);    // LN scratch
  float* QKV  = alloc(3145728);    // [2048,1536]
  float* AO   = alloc(1048576);
  float* FCb  = alloc(4194304);    // [2048,2048]
  float* GMb  = alloc(16384);
  float* GFb  = alloc(16384);
  float* INA  = alloc(32);
  float* INB  = alloc(32);

  auto gemm = [&](const float* A, int lda, const float* Bm, float* C,
                  int M, int N, int K, const float* sc, float smul,
                  const float* bi, const float* ad, int fl) {
    dim3 grid((M + BMT - 1) / BMT, (N + BNT - 1) / BNT);
    gemm_bf16_wmma<<<grid, dim3(256), 0, stream>>>(A, lda, Bm, C, M, N, K, sc, smul, bi, ad, fl);
  };
  auto conv3 = [&](const float* X, int nchw, const float* w_oihw, int O, int I,
                   int Nn, int Hh, int Ww, int strd, float* Y,
                   const float* g, const float* b) {
    int tot = O * I * 9;
    k_wreorder<<<(tot + 255) / 256, 256, 0, stream>>>(w_oihw, WR, O, I, 3, 3);
    int OH = (Hh + 2 - 3) / strd + 1, OW = (Ww + 2 - 3) / strd + 1;
    int M = Nn * OH * OW;
    dim3 grid((M + BMT - 1) / BMT, (O + BNT - 1) / BNT);
    conv_bf16_wmma<<<grid, dim3(256), 0, stream>>>(X, nchw, WR, Y, Nn, Hh, Ww, I, O,
                                                   3, 3, strd, 1, OH, OW,
                                                   g, BN_SCALE, b, nullptr, 1);
  };
  auto pool2 = [&](const float* in, float* out, int Nn, int H, int Wd, int C) {
    size_t tot = (size_t)Nn * (H / 2) * (Wd / 2) * C;
    k_avgpool2<<<(int)((tot + 255) / 256), 256, 0, stream>>>(in, out, Nn, H, Wd, C);
  };

  // ===== Param indices: jax tree_flatten order (alphabetical dict keys) =====
  // 0 image, 1 graph, 2 adj
  // 3..11  attnpool: cb cw kb kw pos qb qw vb vw
  // 12..83 blocks[6]: fc_b fc_w in_b in_w ln1b ln1g ln2b ln2g out_b out_w pj_b pj_w
  // 84..167 layers (stages; block w/ downsample = 12, else 9)
  // 168 lnfb, 169 lnfg, 170..178 stem: b1 b2 b3 g1 g2 g3 w1 w2 w3, 179 tok

  // ---- stem ----
  const int sB = 170;
  conv3(image, 1, F(sB + 6), 32, 3, 32, 224, 224, 2, ACT0, F(sB + 3), F(sB + 0));
  conv3(ACT0, 0, F(sB + 7), 32, 32, 32, 112, 112, 1, ACT1, F(sB + 4), F(sB + 1));
  conv3(ACT1, 0, F(sB + 8), 64, 32, 32, 112, 112, 1, ACT0, F(sB + 5), F(sB + 2));
  pool2(ACT0, ACT1, 32, 112, 112, 64);

  // ---- bottleneck stages ----
  float* cur = ACT1; float* other = ACT0;
  int Hh = 56, Wc = 56, inc = 64;
  int pidx = 84;
  const int planes_arr[4] = {64, 128, 256, 512};
  for (int s = 0; s < 4; ++s) {
    int planes = planes_arr[s];
    for (int bi = 0; bi < 2; ++bi) {
      int strd = (s > 0 && bi == 0) ? 2 : 1;
      bool has_wd = (strd > 1) || (inc != planes * 4);
      const float *b1, *b2, *b3, *bd = nullptr, *g1, *g2, *g3, *gd = nullptr;
      const float *w1, *w2, *w3, *wdp = nullptr;
      if (has_wd) {
        b1 = F(pidx + 0); b2 = F(pidx + 1); b3 = F(pidx + 2); bd = F(pidx + 3);
        g1 = F(pidx + 4); g2 = F(pidx + 5); g3 = F(pidx + 6); gd = F(pidx + 7);
        w1 = F(pidx + 8); w2 = F(pidx + 9); w3 = F(pidx + 10); wdp = F(pidx + 11);
        pidx += 12;
      } else {
        b1 = F(pidx + 0); b2 = F(pidx + 1); b3 = F(pidx + 2);
        g1 = F(pidx + 3); g2 = F(pidx + 4); g3 = F(pidx + 5);
        w1 = F(pidx + 6); w2 = F(pidx + 7); w3 = F(pidx + 8);
        pidx += 9;
      }
      int M0 = 32 * Hh * Wc;
      gemm(cur, inc, w1, TMP0, M0, planes, inc, g1, BN_SCALE, b1, nullptr, 1);  // 1x1 + BN + ReLU
      conv3(TMP0, 0, w2, planes, planes, 32, Hh, Wc, 1, TMP1, g2, b2);          // 3x3 + BN + ReLU
      float* optr = TMP1;
      int oh = Hh, ow = Wc;
      if (strd > 1) { oh = Hh / 2; ow = Wc / 2; pool2(TMP1, TMP0, 32, Hh, Wc, planes); optr = TMP0; }
      int Mo = 32 * oh * ow;
      const float* ires;
      if (has_wd) {
        const float* ip = cur;
        if (strd > 1) { pool2(cur, TMP1, 32, Hh, Wc, inc); ip = TMP1; }
        gemm(ip, inc, wdp, other, Mo, planes * 4, inc, gd, BN_SCALE, bd, nullptr, 0);
        ires = other;
      } else {
        ires = cur;
      }
      gemm(optr, planes, w3, other, Mo, planes * 4, planes, g3, BN_SCALE, b3, ires, 1);
      float* t = cur; cur = other; other = t;
      Hh = oh; Wc = ow; inc = planes * 4;
    }
  }

  // ---- attention pool (cur = [32,7,7,2048] NHWC = [32,49,2048]) ----
  {
    size_t tot = (size_t)32 * 50 * 2048;
    k_attn_tokens<<<(int)((tot + 255) / 256), 256, 0, stream>>>(cur, F(7), TCAT, 32, 2048);
  }
  gemm(TCAT, 50 * 2048, F(9), QP, 32, 2048, 2048, nullptr, 1.f, F(8), nullptr, 0);  // q (token0 rows)
  gemm(TCAT, 2048, F(6), KP, 1600, 2048, 2048, nullptr, 1.f, F(5), nullptr, 0);     // k
  gemm(TCAT, 2048, F(11), VP, 1600, 2048, 2048, nullptr, 1.f, F(10), nullptr, 0);   // v
  k_attnpool<<<32 * 8, 256, 0, stream>>>(QP, KP, VP, ATTO);
  gemm(ATTO, 2048, F(4), IMGF, 32, 512, 2048, nullptr, 1.f, F(3), nullptr, 0);      // c_proj

  // ---- graph transformer ----
  k_embed<<<(32 * 64 * 512 + 255) / 256, 256, 0, stream>>>(graphi, F(179), Gb, 32 * 64, 512);
  for (int t = 0; t < 6; ++t) {
    int bp = 12 + 12 * t;
    const float *fc_b = F(bp + 0), *fc_w = F(bp + 1), *in_b = F(bp + 2), *in_w = F(bp + 3);
    const float *ln1b = F(bp + 4), *ln1g = F(bp + 5), *ln2b = F(bp + 6), *ln2g = F(bp + 7);
    const float *out_b = F(bp + 8), *out_w = F(bp + 9), *pj_b = F(bp + 10), *pj_w = F(bp + 11);
    k_layernorm<<<2048, 256, 0, stream>>>(Gb, Hb, ln1g, ln1b, 512);
    gemm(Hb, 512, in_w, QKV, 2048, 1536, 512, nullptr, 1.f, in_b, nullptr, 0);
    k_graph_attn<<<32 * 8 * 64, 64, 0, stream>>>(QKV, adj, AO);
    gemm(AO, 512, out_w, Gb, 2048, 512, 512, nullptr, 1.f, out_b, Gb, 0);           // +residual
    k_layernorm<<<2048, 256, 0, stream>>>(Gb, Hb, ln2g, ln2b, 512);
    gemm(Hb, 512, fc_w, FCb, 2048, 2048, 512, nullptr, 1.f, fc_b, nullptr, 2);      // quickGELU
    gemm(FCb, 2048, pj_w, Gb, 2048, 512, 2048, nullptr, 1.f, pj_b, Gb, 0);          // +residual
  }
  k_mean_rows<<<(32 * 512 + 255) / 256, 256, 0, stream>>>(Gb, GMb, 32, 64, 512);
  k_layernorm<<<32, 256, 0, stream>>>(GMb, GFb, F(169), F(168), 512);

  // ---- fused normalize + 32x32 similarity ----
  k_row_inv_norm<<<32, 256, 0, stream>>>(IMGF, INA, 512);
  k_row_inv_norm<<<32, 256, 0, stream>>>(GFb, INB, 512);
  k_cosine<<<1, 1024, 0, stream>>>(IMGF, INA, GFb, INB, (float*)d_out);
}
